// NWSparseKernelRegression_33715493273880
// MI455X (gfx1250) — matrix operand order repaired and checked by reading, not compiled, not run on hardware
//
#include <hip/hip_runtime.h>

typedef __attribute__((ext_vector_type(16))) _Float16 v16h;
typedef __attribute__((ext_vector_type(8)))  float    v8f;

#define N_PTS  8192
#define VCOLS  8
#define TILES  (N_PTS / 16)   // 512 query tiles
#define CHUNKS (N_PTS / 32)   // 256 key chunks

// ---------------------------------------------------------------------------
// Prep: pad keys to float4{x,y,z,|k|^2} and pre-pack values into per-lane
// f16 B-fragments for v_wmma_f32_16x16x32_f16.
// B is 32x16: cols 0..7 = values, col 8 = 1.0 (den column), cols 9..15 = 0.
// Dense 16-bit B layout (wave32): lanes 0-15 hold K=0..7,16..23 (2 K per VGPR),
// lanes 16-31 hold K=8..15,24..31.
// ---------------------------------------------------------------------------
__global__ void nw_prep(const float* __restrict__ keys,
                        const float* __restrict__ values,
                        float4* __restrict__ keyspad,
                        _Float16* __restrict__ bpack) {
    int tid = blockIdx.x * blockDim.x + threadIdx.x;
    if (tid < N_PTS) {
        float kx = keys[3 * tid + 0];
        float ky = keys[3 * tid + 1];
        float kz = keys[3 * tid + 2];
        keyspad[tid] = make_float4(kx, ky, kz, kx * kx + ky * ky + kz * kz);
    } else {
        int t    = tid - N_PTS;        // 0..8191 : (chunk, lane)
        int chunk = t >> 5;
        int lane  = t & 31;
        int n = lane & 15;
        int h = lane >> 4;
        v16h b;
#pragma unroll
        for (int j = 0; j < 16; ++j) {
            int k = chunk * 32 + 8 * h + j + 8 * (j >> 3);
            float x = (n < VCOLS) ? values[k * VCOLS + n]
                                  : ((n == VCOLS) ? 1.0f : 0.0f);
            b[j] = (_Float16)x;
        }
        *(v16h*)(bpack + (size_t)t * 16) = b;
    }
}

// ---------------------------------------------------------------------------
// Main: one wave owns a 16-query tile and a segment of the key dimension.
// Per 32-key chunk: compute 16 kernel weights per lane directly in the
// f16 A-fragment layout (16-bit A 16x32: lane m=L&15; h=L>>4 selects
// K = 8h..8h+7 in v0..3 and K = 16+8h..16+8h+7 in v4..7), then one
// v_wmma_f32_16x16x32_f16 accumulates num (cols 0-7) and den (col 8).
// Weights are computed scaled by 3 (num/den is invariant to uniform
// positive scaling, and relu(3w)=3*relu(w)), saving a multiply per pair.
// v_sin/v_cos take revolutions: sin(2*pi*cdist) == v_sin(cdist).
// Wave-uniform control values are forced into SGPRs via readfirstlane so
// the chunk loop is a scalar loop and EXEC stays all-1s for the WMMA.
// ---------------------------------------------------------------------------
__global__ void nw_main(const float* __restrict__ queries,
                        const float* __restrict__ kernelLen,
                        const float4* __restrict__ keyspad,
                        const _Float16* __restrict__ bpack,
                        float* __restrict__ part,
                        int seg_count) {
    const int lane  = threadIdx.x & 31;
    const int gwave = __builtin_amdgcn_readfirstlane(
        (blockIdx.x * blockDim.x + threadIdx.x) >> 5);
    const int tile  = gwave % TILES;
    const int seg   = gwave / TILES;

    const int m = tile * 16 + (lane & 15);
    const int h = lane >> 4;

    const float qx = queries[3 * m + 0];
    const float qy = queries[3 * m + 1];
    const float qz = queries[3 * m + 2];
    const float qsq = qx * qx + qy * qy + qz * qz;
    const float ax = -2.0f * qx, ay = -2.0f * qy, az = -2.0f * qz;
    const float invLen = 1.0f / kernelLen[0];

    const int chunks_per_seg = __builtin_amdgcn_readfirstlane(CHUNKS / seg_count);
    const int c0   = __builtin_amdgcn_readfirstlane(seg * chunks_per_seg);
    const int cend = c0 + chunks_per_seg;

    v8f acc = {};
    for (int chunk = c0; chunk < cend; ++chunk) {
        const float4* kp = keyspad + chunk * 32 + 8 * h;
        __builtin_prefetch(kp + 32, 0, 1);   // global_prefetch_b8 for next chunk

        v16h a;
#pragma unroll
        for (int g = 0; g < 2; ++g) {
#pragma unroll
            for (int i = 0; i < 8; ++i) {
                float4 kv = kp[g * 16 + i];
                float d2 = qsq + kv.w;
                d2 = __builtin_fmaf(ax, kv.x, d2);
                d2 = __builtin_fmaf(ay, kv.y, d2);
                d2 = __builtin_fmaf(az, kv.z, d2);
                d2 = fmaxf(d2, 0.0f);
                float cd   = __builtin_amdgcn_sqrtf(d2) * invLen;
                float cosv = __builtin_amdgcn_cosf(cd);   // cos(2*pi*cd)
                float sinv = __builtin_amdgcn_sinf(cd);   // sin(2*pi*cd)
                // 3x-scaled weight: (2+cos)(1-cd) + sin * 3/(2*pi)
                float w = __builtin_fmaf(sinv, 0.477464829275686f,
                                         (2.0f + cosv) * (1.0f - cd));
                w = fmaxf(w, 0.0f);
                a[g * 8 + i] = (_Float16)w;
            }
        }
        v16h b = *(const v16h*)(bpack + (size_t)(chunk * 32 + lane) * 16);
        acc = __builtin_amdgcn_wmma_f32_16x16x32_f16(
            /*neg_a=*/false, a, /*neg_b=*/false, b,
            /*c_mod=*/(short)0, acc, /*reuse_a=*/false, /*reuse_b=*/false);
    }

    // Dump the 16x16 f32 accumulator in native layout: VGPR r, lane l.
    float* dst = part + (size_t)(seg * TILES + tile) * 256;
#pragma unroll
    for (int r = 0; r < 8; ++r) dst[r * 32 + lane] = acc[r];
}

// ---------------------------------------------------------------------------
// Reduce: sum per-segment partials, divide num by den (col 8), add 1e-6.
// C layout: VGPR r / lane l: rows 0-7 on lanes 0-15, rows 8-15 on 16-31.
// ---------------------------------------------------------------------------
__global__ void nw_reduce(const float* __restrict__ part,
                          float* __restrict__ out,
                          int seg_count) {
    int t = blockIdx.x * blockDim.x + threadIdx.x;     // 0..65535
    if (t >= N_PTS * VCOLS) return;
    int m = t >> 3, j = t & 7;
    int tile = m >> 4, mr = m & 15;
    int half = mr >> 3, v = mr & 7;
    int lane_n = 16 * half + j;
    int lane_d = 16 * half + 8;
    float num = 0.0f, den = 0.0f;
    for (int s = 0; s < seg_count; ++s) {
        const float* p = part + (size_t)(s * TILES + tile) * 256 + v * 32;
        num += p[lane_n];
        den += p[lane_d];
    }
    out[t] = num / den + 1e-6f;
}

// ---------------------------------------------------------------------------
extern "C" void kernel_launch(void* const* d_in, const int* in_sizes, int n_in,
                              void* d_out, int out_size, void* d_ws, size_t ws_size,
                              hipStream_t stream) {
    const float* queries   = (const float*)d_in[0];   // [8192,3]
    const float* keys      = (const float*)d_in[1];   // [8192,3]
    const float* values    = (const float*)d_in[2];   // [8192,8]
    const float* kernelLen = (const float*)d_in[3];   // [1]
    float* out = (float*)d_out;                       // [8192,8]

    char* ws = (char*)d_ws;
    const size_t keyspad_bytes = (size_t)N_PTS * 16;          // 128 KB
    const size_t bpack_bytes   = (size_t)N_PTS * 32;          // 256 KB
    float4*   keyspad = (float4*)ws;
    _Float16* bpack   = (_Float16*)(ws + keyspad_bytes);
    float*    part    = (float*)(ws + keyspad_bytes + bpack_bytes);
    const size_t base = keyspad_bytes + bpack_bytes;

    // Pick the largest key-dimension split that fits in the workspace (occupancy knob).
    int seg = 16;
    while (seg > 1 && base + (size_t)seg * TILES * 256 * sizeof(float) > ws_size)
        seg >>= 1;

    nw_prep<<<(2 * N_PTS) / 256, 256, 0, stream>>>(keys, values, keyspad, bpack);

    int waves  = seg * TILES;           // up to 8192 waves
    int blocks = waves / 8;             // 8 waves (256 threads) per block
    nw_main<<<blocks, 256, 0, stream>>>(queries, kernelLen, keyspad, bpack, part, seg);

    nw_reduce<<<(N_PTS * VCOLS) / 256, 256, 0, stream>>>(part, out, seg);
}
